// TempoAttention_14491219657086
// MI455X (gfx1250) — compile-verified, matrix operand order checked by reading, not run on hardware
//
#include <hip/hip_runtime.h>
#include <stdint.h>

// ---------------------------------------------------------------------------
// Fused attention for MI455X (gfx1250, wave32, WMMA bf16 f32-accum,
// async global->LDS double-buffered staging on ASYNCcnt via inline asm)
// N=8, T=1024, D=128, H=8
// ---------------------------------------------------------------------------

typedef __bf16 bf16_t;
typedef bf16_t v16bf __attribute__((ext_vector_type(16)));
typedef float  v8f   __attribute__((ext_vector_type(8)));

constexpr int N  = 8;
constexpr int T  = 1024;
constexpr int D  = 128;
constexpr int H  = 8;
constexpr int HD = H * D;                       // 1024
constexpr float SCALE = 0.08838834764831845f;   // 1/sqrt(128)

#ifndef __has_builtin
#define __has_builtin(x) 0
#endif

// ---- CDNA5 counter / fence helpers ----------------------------------------
__device__ __forceinline__ void lds_wave_fence() {
  __builtin_amdgcn_wave_barrier();
  asm volatile("s_wait_dscnt 0x0" ::: "memory");
  __builtin_amdgcn_wave_barrier();
}
__device__ __forceinline__ void wait_async0() {
#if __has_builtin(__builtin_amdgcn_s_wait_asynccnt)
  __builtin_amdgcn_s_wait_asynccnt(0);
#else
  asm volatile("s_wait_asynccnt 0x0" ::: "memory");
#endif
  __builtin_amdgcn_wave_barrier();
}
__device__ __forceinline__ void wait_async8() {
#if __has_builtin(__builtin_amdgcn_s_wait_asynccnt)
  __builtin_amdgcn_s_wait_asynccnt(8);
#else
  asm volatile("s_wait_asynccnt 0x8" ::: "memory");
#endif
  __builtin_amdgcn_wave_barrier();
}

// Async 16-byte global -> LDS copy (GLOBAL_LOAD_ASYNC_TO_LDS_B128, ASYNCcnt).
// LDS operand is the wave-relative LDS byte offset = low 32 bits of the
// generic pointer to a __shared__ object.
__device__ __forceinline__ void async_cp16(const void* g, void* l) {
  uint32_t lds_off = (uint32_t)(uintptr_t)l;
  uint64_t gaddr   = (uint64_t)(uintptr_t)g;
  asm volatile("global_load_async_to_lds_b128 %0, %1, off"
               :: "v"(lds_off), "v"(gaddr)
               : "memory");
}

// Stage a 4 KB tile (16x128 bf16) with one wave (8 x b128 per lane).
__device__ __forceinline__ void stage4k_wave(const bf16_t* g, bf16_t* l, int lane) {
#pragma unroll
  for (int i = 0; i < 8; ++i)
    async_cp16(g + (size_t)(lane + i * 32) * 8, l + (size_t)(lane + i * 32) * 8);
}
// Stage an 8 KB tile (32x128 bf16) with the whole 256-thread block.
__device__ __forceinline__ void stage8k_block(const bf16_t* g, bf16_t* l, int tid) {
  async_cp16(g + (size_t)tid * 8, l + (size_t)tid * 8);
  async_cp16(g + (size_t)(tid + 256) * 8, l + (size_t)(tid + 256) * 8);
}
// Stage a 4 KB tile with the whole block (1 x b128 per thread).
__device__ __forceinline__ void stage4k_block(const bf16_t* g, bf16_t* l, int tid) {
  async_cp16(g + (size_t)tid * 8, l + (size_t)tid * 8);
}

// ---- WMMA fragment helpers (CDNA5 ISA 7.12.2 layouts) ---------------------
// K index of element e (0..15) of a 16-bit A/B fragment.
__device__ __forceinline__ int frag_k(int e, int hi) {
  int v = e >> 1;
  return ((v & 4) << 2) | ((v & 3) << 1) | (hi << 3) | (e & 1);
}
// A fragment (M = lane%16) from row-major [16][128] bf16 buffer.
__device__ __forceinline__ v16bf frag_A(const bf16_t* buf, int lane) {
  int r = lane & 15, hi = lane >> 4;
  v16bf f;
#pragma unroll
  for (int e = 0; e < 16; ++e) f[e] = buf[r * 128 + frag_k(e, hi)];
  return f;
}
// B fragment where B[k][c] = buf[c][k]  (transposed tile: K^T, W^T).
__device__ __forceinline__ v16bf frag_B_T(const bf16_t* buf, int lane) {
  return frag_A(buf, lane);
}
// B fragment from row-major [32][128] buffer: B[k][c] = buf[k][colbase+c].
__device__ __forceinline__ v16bf frag_B_rm(const bf16_t* buf, int colbase, int lane) {
  int c = lane & 15, hi = lane >> 4;
  v16bf f;
#pragma unroll
  for (int e = 0; e < 16; ++e) f[e] = buf[frag_k(e, hi) * 128 + colbase + c];
  return f;
}
__device__ __forceinline__ v8f wmma_bf16(v16bf a, v16bf b, v8f c) {
  return __builtin_amdgcn_wmma_f32_16x16x32_bf16(false, a, false, b, (short)0, c,
                                                 false, false);
}

// ---------------------------------------------------------------------------
// Kernel 1: Q/K/V projections. grid=(NT/16, 3), block=256 (8 waves).
// ---------------------------------------------------------------------------
__global__ __launch_bounds__(256) void tempo_proj_kernel(
    const float* __restrict__ x,
    const float* __restrict__ Wq, const float* __restrict__ bq,
    const float* __restrict__ Wk, const float* __restrict__ bk,
    const float* __restrict__ Wv, const float* __restrict__ bv,
    bf16_t* __restrict__ qws, bf16_t* __restrict__ kws, bf16_t* __restrict__ vws) {
  int mt = blockIdx.x, sel = blockIdx.y;
  const float* W    = (sel == 0) ? Wq : (sel == 1) ? Wk : Wv;
  const float* bias = (sel == 0) ? bq : (sel == 1) ? bk : bv;
  bf16_t*      dst  = (sel == 0) ? qws : (sel == 1) ? kws : vws;

  __shared__ bf16_t xs[16 * 128];
  __shared__ bf16_t Wbuf[8][16 * 128];

  const float* xg = x + (size_t)mt * 16 * 128;
  for (int i = threadIdx.x; i < 16 * 128; i += 256) xs[i] = (bf16_t)xg[i];
  __syncthreads();

  int wave = threadIdx.x >> 5, lane = threadIdx.x & 31, hi = lane >> 4;
  for (int nt = wave; nt < HD / 16; nt += 8) {
    const float* wg = W + (size_t)nt * 16 * 128;
    for (int i = lane; i < 16 * 128; i += 32) Wbuf[wave][i] = (bf16_t)wg[i];
    lds_wave_fence();

    v8f acc = {};
#pragma unroll
    for (int kk = 0; kk < 4; ++kk)
      acc = wmma_bf16(frag_A(xs + kk * 32, lane),
                      frag_B_T(&Wbuf[wave][kk * 32], lane), acc);

    int o = nt * 16 + (lane & 15);
    float bb = bias[o];
    int h = o >> 7, dd = o & 127;
#pragma unroll
    for (int i = 0; i < 8; ++i) {
      int mrow = mt * 16 + i + (hi << 3);
      int n = mrow >> 10, t = mrow & 1023;
      dst[(((size_t)n * H + h) * T + t) * D + dd] = (bf16_t)(acc[i] + bb);
    }
    lds_wave_fence();
  }
}

// ---------------------------------------------------------------------------
// Kernel 2: fused scores -> softmax -> att -> P@V.
// grid=(T/16, H, N), block=256 (8 waves). Score strip in LDS; K and V tiles
// double-buffered via async global->LDS (ASYNCcnt).
// ---------------------------------------------------------------------------
__global__ __launch_bounds__(256) void tempo_attn_kernel(
    const bf16_t* __restrict__ qws, const bf16_t* __restrict__ kws,
    const bf16_t* __restrict__ vws,
    const unsigned char* __restrict__ mask, const float* __restrict__ aw,
    float* __restrict__ outp, float* __restrict__ attp) {
  int mt = blockIdx.x, h = blockIdx.y, n = blockIdx.z;

  __shared__ float  Ss[16 * 1024];          // 64 KB score strip
  __shared__ bf16_t Qs[16 * 128];           // 4 KB
  __shared__ bf16_t Kbuf[8][2][16 * 128];   // 64 KB per-wave double-buffered K
  __shared__ bf16_t Vs[2][32 * 128];        // 16 KB double-buffered V
  __shared__ float  red[16][16];
  __shared__ float  rowmax16[16];
  __shared__ float  rowinv16[16];

  int tid = threadIdx.x, wave = tid >> 5, lane = tid & 31, hi = lane >> 4;
  size_t nh = (size_t)n * H + h;
  const bf16_t* kbase = kws + nh * T * D;

  // Stage Q strip + first K tile, drain ASYNCcnt, block-sync.
  stage4k_block(qws + (nh * T + (size_t)mt * 16) * D, Qs, tid);
  stage4k_wave(kbase + (size_t)wave * 16 * D, Kbuf[wave][0], lane);
  wait_async0();
  __syncthreads();

  // ---- scores: S = (Q K^T) * scale * attention_weights, masked -> -inf ----
  for (int jj = 0; jj < 8; ++jj) {
    int j = wave + jj * 8;
    int cur = jj & 1;
    bool more = (jj + 1) < 8;
    if (more)
      stage4k_wave(kbase + (size_t)(j + 8) * 16 * D, Kbuf[wave][cur ^ 1], lane);
    if (more) wait_async8(); else wait_async0();   // in-order: cur tile landed

    v8f acc = {};
#pragma unroll
    for (int kk = 0; kk < 4; ++kk)
      acc = wmma_bf16(frag_A(Qs + kk * 32, lane),
                      frag_B_T(&Kbuf[wave][cur][kk * 32], lane), acc);

    int c = j * 16 + (lane & 15);
    bool msk = mask[n * T + c] != 0;
    const float* awc = aw + (nh * T + (size_t)mt * 16) * T + c;
#pragma unroll
    for (int i = 0; i < 8; ++i) {
      int m = i + (hi << 3);
      float s = acc[i] * SCALE * awc[(size_t)m * T];
      if (msk) s = -__builtin_inff();
      Ss[m * 1024 + c] = s;
    }
  }
  __syncthreads();

  // ---- softmax over each of the 16 rows (1024 cols) ----
  int r = tid >> 4, cp = tid & 15;
  float lmax = -__builtin_inff();
  for (int i = 0; i < 64; ++i) lmax = fmaxf(lmax, Ss[r * 1024 + cp + i * 16]);
  red[r][cp] = lmax;
  __syncthreads();
  if (tid < 16) {
    float m2 = red[tid][0];
    for (int i = 1; i < 16; ++i) m2 = fmaxf(m2, red[tid][i]);
    rowmax16[tid] = m2;
  }
  __syncthreads();
  float rm = rowmax16[r];
  float lsum = 0.f;
  for (int i = 0; i < 64; ++i) {
    int idx = r * 1024 + cp + i * 16;
    float e = __expf(Ss[idx] - rm);
    Ss[idx] = e;
    lsum += e;
  }
  red[r][cp] = lsum;
  __syncthreads();
  if (tid < 16) {
    float s2 = 0.f;
    for (int i = 0; i < 16; ++i) s2 += red[tid][i];
    rowinv16[tid] = 1.0f / s2;
  }
  __syncthreads();

  // Prefetch first V tile before the att write so loads overlap the stores.
  const bf16_t* vg = vws + nh * T * D;
  stage8k_block(vg, Vs[0], tid);

  // ---- write normalized att (coalesced, fire-and-forget on STOREcnt) ----
  {
    float* ag = attp + (nh * T + (size_t)mt * 16) * T;
    for (int l = tid; l < 16 * 1024; l += 256) {
      int rr = l >> 10, cc = l & 1023;
      ag[(size_t)rr * T + cc] = Ss[l] * rowinv16[rr];
    }
  }

  // ---- P @ V : wave w produces out cols [w*16, w*16+16) ----
  v8f o = {};
  int ra = lane & 15;
  float rinv = rowinv16[ra];  // fold 1/sum into the bf16 A-frag conversion
  for (int kt = 0; kt < T / 32; ++kt) {
    int cur = kt & 1;
    wait_async0();            // cur tile (issued one iteration ago) landed
    __syncthreads();          // all waves see cur; prev reads of nxt done
    if (kt + 1 < T / 32)
      stage8k_block(vg + (size_t)(kt + 1) * 32 * 128, Vs[cur ^ 1], tid);
    v16bf a;
#pragma unroll
    for (int e = 0; e < 16; ++e)
      a[e] = (bf16_t)(Ss[ra * 1024 + kt * 32 + frag_k(e, hi)] * rinv);
    v16bf b = frag_B_rm(Vs[cur], wave * 16, lane);
    o = wmma_bf16(a, b, o);
  }
#pragma unroll
  for (int i = 0; i < 8; ++i) {
    int m = i + (hi << 3);
    outp[((size_t)n * T + mt * 16 + m) * HD + h * D + wave * 16 + (lane & 15)] = o[i];
  }
}

// ---------------------------------------------------------------------------
// Kernel 3: out_agg[n][c] = mean_t out[n][t][c]
// ---------------------------------------------------------------------------
__global__ __launch_bounds__(256) void tempo_agg_kernel(const float* __restrict__ outp,
                                                        float* __restrict__ aggp) {
  int g = blockIdx.x * 256 + threadIdx.x;
  int n = g >> 10, c = g & 1023;
  float s = 0.f;
  for (int t = 0; t < T; ++t) s += outp[((size_t)n * T + t) * HD + c];
  aggp[g] = s * (1.0f / T);
}

// ---------------------------------------------------------------------------
extern "C" void kernel_launch(void* const* d_in, const int* in_sizes, int n_in,
                              void* d_out, int out_size, void* d_ws, size_t ws_size,
                              hipStream_t stream) {
  (void)in_sizes; (void)n_in; (void)out_size; (void)ws_size;
  const float*         x    = (const float*)d_in[0];
  const unsigned char* mask = (const unsigned char*)d_in[1];  // jnp.bool_ = 1B
  const float*         aw   = (const float*)d_in[2];
  const float* Wq = (const float*)d_in[3];
  const float* bq = (const float*)d_in[4];
  const float* Wk = (const float*)d_in[5];
  const float* bk = (const float*)d_in[6];
  const float* Wv = (const float*)d_in[7];
  const float* bv = (const float*)d_in[8];

  // d_out = [ out (N*T*HD) | out_agg (N*HD) | att (N*H*T*T) ]
  float* outp = (float*)d_out;
  float* aggp = outp + (size_t)N * T * HD;
  float* attp = aggp + (size_t)N * HD;

  // workspace: Q,K,V as bf16 [N,H,T,D]  (3 * 16 MB)
  bf16_t* qws = (bf16_t*)d_ws;
  bf16_t* kws = qws + (size_t)N * H * T * D;
  bf16_t* vws = kws + (size_t)N * H * T * D;

  tempo_proj_kernel<<<dim3((N * T) / 16, 3), 256, 0, stream>>>(
      x, Wq, bq, Wk, bk, Wv, bv, qws, kws, vws);
  tempo_attn_kernel<<<dim3(T / 16, H, N), 256, 0, stream>>>(
      qws, kws, vws, mask, aw, outp, attp);
  tempo_agg_kernel<<<(N * HD) / 256, 256, 0, stream>>>(outp, aggp);
}
